// SplendorNNet_18605798326416
// MI455X (gfx1250) — matrix operand order repaired and checked by reading, not compiled, not run on hardware
//
#include <hip/hip_runtime.h>
#include <hip/hip_bf16.h>

#define BATCH 32768
#define NBV 56
#define VD 7

typedef __attribute__((ext_vector_type(16))) _Float16 v16h;
typedef __attribute__((ext_vector_type(8)))  float    v8f;
typedef __attribute__((ext_vector_type(4)))  unsigned int v4u;

union Frag16 { v16h h; v4u q[2]; };

// ---- weight-fragment pool offsets (f16 elements) ----
#define OFF_W1A 0        // 256x64
#define OFF_W1B 16384    // 256x256
#define OFF_GP1 81920    // 256x256 (noff=8, koff=32)
#define OFF_W3  147456   // 128x256
#define OFF_W4  180224   // 256x704
#define OFF_GP4 360448   // 256x256 (noff=8, koff=16)
#define OFF_W5  425984   // 128x256
#define OFF_GP5 458752   // 128x128 (noff=8, koff=16)
#define OFF_PI1 475136   // 128x128
#define OFF_PI2 491520   // 96x128
#define OFF_V1  503808   // 128x128
#define OFF_SD1 520192   // 128x128
#define OFF_SD2 536576   // 32x128
#define WF_ELEMS 540672
#define WF_BYTES (WF_ELEMS * 2)

// Convert f32 weight (N,K) row-major into f16 WMMA B-fragment layout with
// optional front padding in N (noff) and K (koff); out-of-range -> 0.
__global__ __launch_bounds__(256) void wconv(const float* __restrict__ W,
                                             _Float16* __restrict__ dst,
                                             int N, int K, int Npad, int Kpad,
                                             int noff, int koff) {
  int i = blockIdx.x * blockDim.x + threadIdx.x;
  if (i >= Npad * Kpad) return;
  int n = i / Kpad, k = i - n * Kpad;
  float v = 0.f;
  int nn = n - noff, kk = k - koff;
  if (nn >= 0 && nn < N && kk >= 0 && kk < K) v = W[nn * K + kk];
  int nt = n >> 4, nl = n & 15, kt = k >> 5, kin = k & 31;
  int lane = nl + (kin & 16);
  int vv = (kin & 15) >> 1;
  long idx = ((long)(nt * (Kpad >> 5) + kt) * 32 + lane) * 16 + vv * 2 + (k & 1);
  dst[idx] = (_Float16)v;
}

// A fragment (16x32 f16) from row-major LDS tile: lanes 0-15 rows, K split by lane half
__device__ inline v16h load_a_frag(const _Float16* __restrict__ aRow, int kbase, int half) {
  const _Float16* p = aRow + kbase + (half << 4);
  Frag16 f;
  f.q[0] = *(const v4u*)(p);
  f.q[1] = *(const v4u*)(p + 8);
  return f.h;
}

// B fragment: contiguous 1KB blocks per (ntile,kstep); two coalesced b128 per lane
__device__ inline v16h load_b_frag(const _Float16* __restrict__ wf, int nt, int kt, int nK, int lane) {
  const v4u* p = (const v4u*)(wf + ((long)(nt * nK + kt) * 32 + lane) * 16);
  Frag16 f;
  f.q[0] = p[0];
  f.q[1] = p[1];
  return f.h;
}

// MODE: 0 bias+relu->f16 LDS | 1 bias->f16 LDS | 2 bias->f32 LDS | 3 bias+BN+relu->f16 LDS
//       4 bias+relu->f16 GLOBAL.   MT = number of 16-row M subtiles (1 or 2).
template<int MODE, int MT>
__device__ inline void gemm_t(const _Float16* __restrict__ aLds, int astride,
                              const _Float16* __restrict__ wf,
                              const float* __restrict__ bias,
                              int nK, int Ntiles,
                              _Float16* __restrict__ oH, int ostride,
                              float* __restrict__ oF, int ofstride,
                              const float* __restrict__ sc7, const float* __restrict__ sh7,
                              long mbase, _Float16* __restrict__ gOut, int gstride) {
  const int lane = threadIdx.x & 31;
  const int wave = threadIdx.x >> 5;
  const int half = lane >> 4;
  const _Float16* aRow0 = aLds + (lane & 15) * astride;
  for (int nt = wave; nt < Ntiles; nt += 4) {
    v8f acc[MT];
#pragma unroll
    for (int s = 0; s < MT; ++s) acc[s] = (v8f){};
    for (int kt = 0; kt < nK; ++kt) {
      v16h b = load_b_frag(wf, nt, kt, nK, lane);
#pragma unroll
      for (int s = 0; s < MT; ++s) {
        v16h a = load_a_frag(aRow0 + (s << 4) * astride, kt << 5, half);
        acc[s] = __builtin_amdgcn_wmma_f32_16x16x32_f16(
            false, a, false, b, (short)0, acc[s], false, false);
      }
    }
    const int n = (nt << 4) + (lane & 15);
    const float bn_ = bias[n];
#pragma unroll
    for (int s = 0; s < MT; ++s) {
      const int rb = (s << 4) + (half << 3);
#pragma unroll
      for (int v = 0; v < 8; ++v) {
        float x = acc[s][v] + bn_;
        if (MODE == 3) {
          int d = (int)((mbase + rb + v) % 7);
          x = x * sc7[d] + sh7[d];
        }
        if (MODE == 0 || MODE == 3 || MODE == 4) x = fmaxf(x, 0.f);
        if (MODE == 2)      oF[(rb + v) * ofstride + n] = x;
        else if (MODE == 4) gOut[(mbase + rb + v) * (long)gstride + n] = (_Float16)x;
        else                oH[(rb + v) * ostride + n] = (_Float16)x;
      }
    }
  }
}

// ---------------- Stage A: per (batch*7) row, 32 rows per block ----------------
__global__ __launch_bounds__(128) void stageA(
    const float* __restrict__ input, const _Float16* __restrict__ wf,
    const float* __restrict__ b1a, const float* __restrict__ b1b,
    const float* __restrict__ gp1b, const float* __restrict__ b3,
    const float* __restrict__ bng, const float* __restrict__ bnb,
    const float* __restrict__ bnm, const float* __restrict__ bnv,
    _Float16* __restrict__ y3) {
  __shared__ alignas(16) _Float16 actA[32 * 256];
  __shared__ alignas(16) _Float16 actB[32 * 256];
  __shared__ float bias[256];
  __shared__ float sc7[7], sh7[7];
  const int tid = threadIdx.x;
  const long mbase = (long)blockIdx.x * 32;

  if (tid < 7) {
    float s = bng[tid] * rsqrtf(bnv[tid] + 1e-5f);
    sc7[tid] = s;
    sh7[tid] = bnb[tid] - bnm[tid] * s;
  }
  // transpose-load input tile: row m=(b,d), feature c; pad 56->64
  for (int t = tid; t < 32 * 64; t += 128) {
    int r = t >> 6, c = t & 63;
    long m = mbase + r;
    int b = (int)(m / 7), d = (int)(m - (long)b * 7);
    float v = (c < NBV) ? input[((long)b * NBV + c) * VD + d] : 0.f;
    actA[r * 256 + c] = (_Float16)v;
  }
  for (int t = tid; t < 256; t += 128) bias[t] = b1a[t];
  __syncthreads();
  // L1a + batchnorm(d = m%7) + relu
  gemm_t<3, 2>(actA, 256, wf + OFF_W1A, bias, 2, 16, actB, 256, nullptr, 0, sc7, sh7, mbase, nullptr, 0);
  __syncthreads();
  for (int t = tid; t < 256; t += 128) bias[t] = b1b[t];
  __syncthreads();
  gemm_t<0, 2>(actB, 256, wf + OFF_W1B, bias, 8, 16, actA, 256, nullptr, 0, nullptr, nullptr, 0, nullptr, 0);
  __syncthreads();
  for (int t = tid; t < 256; t += 128) bias[t] = (t >= 8) ? gp1b[t - 8] : 0.f;
  __syncthreads();
  // gpool1 dense part: weights front-padded K by 32, N by 8
  gemm_t<0, 2>(actA, 256, wf + OFF_GP1, bias, 8, 16, actB, 256, nullptr, 0, nullptr, nullptr, 0, nullptr, 0);
  __syncthreads();
  // pooled part: 4 groups of 8 -> max/mean into cols 0..7
  for (int t = tid; t < 256; t += 128) {
    int r = t >> 3, o = t & 7, g = o & 3;
    const _Float16* row = actA + r * 256 + g * 8;
    float res;
    if (o < 4) {
      res = (float)row[0];
      for (int i = 1; i < 8; ++i) res = fmaxf(res, (float)row[i]);
    } else {
      float s = 0.f;
      for (int i = 0; i < 8; ++i) s += (float)row[i];
      res = s * 0.125f;
    }
    actB[r * 256 + o] = (_Float16)res;
  }
  for (int t = tid; t < 256; t += 128) bias[t] = (t < 128) ? b3[t] : 0.f;
  __syncthreads();
  gemm_t<4, 2>(actB, 256, wf + OFF_W3, bias, 8, 8, nullptr, 0, nullptr, 0, nullptr, nullptr, mbase, y3, 128);
}

// ---------------- Stage B: per batch, 16 boards per block ----------------
__global__ __launch_bounds__(128) void stageB(
    const _Float16* __restrict__ y3, const _Float16* __restrict__ wf,
    const float* __restrict__ b4, const float* __restrict__ gp4b,
    const float* __restrict__ b5, const float* __restrict__ gp5b,
    const float* __restrict__ pib1, const float* __restrict__ pib2,
    const float* __restrict__ vb1, const float* __restrict__ vw2, const float* __restrict__ vb2,
    const float* __restrict__ sdb1, const float* __restrict__ sdb2,
    const int* __restrict__ valid, float* __restrict__ out) {
  __shared__ alignas(16) _Float16 xcat[16 * 704];
  __shared__ alignas(16) _Float16 actA[16 * 256];
  __shared__ alignas(16) _Float16 actB[16 * 256];
  __shared__ float bias[256];
  __shared__ float piF[16 * 96];
  __shared__ float sdF[16 * 32];
  const int tid = threadIdx.x;
  const long bbase = (long)blockIdx.x * 16;

  // build 704 concat: [max(fc,5x64) | mean(fc) | lc(2x64) | xe(7x64)]
  for (int t = tid; t < 16 * 704; t += 128) {
    int r = t / 704, c = t - r * 704;
    const _Float16* yb = y3 + (bbase + r) * 7 * 128;
    float v;
    if (c < 64) {
      v = (float)yb[c];
      for (int d = 1; d < 5; ++d) v = fmaxf(v, (float)yb[d * 128 + c]);
    } else if (c < 128) {
      int cc = c - 64;
      float s = 0.f;
      for (int d = 0; d < 5; ++d) s += (float)yb[d * 128 + cc];
      v = s * 0.2f;
    } else if (c < 192) v = (float)yb[5 * 128 + (c - 128)];
    else if (c < 256)   v = (float)yb[6 * 128 + (c - 192)];
    else {
      int cc = c - 256;
      v = (float)yb[(cc >> 6) * 128 + 64 + (cc & 63)];
    }
    xcat[r * 704 + c] = (_Float16)v;
  }
  for (int t = tid; t < 256; t += 128) bias[t] = b4[t];
  __syncthreads();
  gemm_t<0, 1>(xcat, 704, wf + OFF_W4, bias, 22, 16, actA, 256, nullptr, 0, nullptr, nullptr, 0, nullptr, 0);
  __syncthreads();
  for (int t = tid; t < 256; t += 128) bias[t] = (t >= 8) ? gp4b[t - 8] : 0.f;
  __syncthreads();
  gemm_t<0, 1>(actA, 256, wf + OFF_GP4, bias, 8, 16, actB, 256, nullptr, 0, nullptr, nullptr, 0, nullptr, 0);
  __syncthreads();
  for (int t = tid; t < 128; t += 128) { } // keep thread layout simple
  for (int t = tid; t < 128; t += 128) { }
  // pooled: 4 groups of 4 from actA cols 0..15
  {
    int r = tid >> 3, o = tid & 7, g = o & 3;
    if (tid < 128) {
      const _Float16* row = actA + r * 256 + g * 4;
      float res;
      if (o < 4) {
        res = (float)row[0];
        for (int i = 1; i < 4; ++i) res = fmaxf(res, (float)row[i]);
      } else {
        float s = 0.f;
        for (int i = 0; i < 4; ++i) s += (float)row[i];
        res = s * 0.25f;
      }
      actB[r * 256 + o] = (_Float16)res;
    }
  }
  for (int t = tid; t < 256; t += 128) bias[t] = (t < 128) ? b5[t] : 0.f;
  __syncthreads();
  gemm_t<0, 1>(actB, 256, wf + OFF_W5, bias, 8, 8, actA, 256, nullptr, 0, nullptr, nullptr, 0, nullptr, 0);
  __syncthreads();
  for (int t = tid; t < 256; t += 128) bias[t] = (t >= 8 && t < 128) ? gp5b[t - 8] : 0.f;
  __syncthreads();
  gemm_t<0, 1>(actA, 256, wf + OFF_GP5, bias, 4, 8, actB, 256, nullptr, 0, nullptr, nullptr, 0, nullptr, 0);
  __syncthreads();
  {
    int r = tid >> 3, o = tid & 7, g = o & 3;
    if (tid < 128) {
      const _Float16* row = actA + r * 256 + g * 4;
      float res;
      if (o < 4) {
        res = (float)row[0];
        for (int i = 1; i < 4; ++i) res = fmaxf(res, (float)row[i]);
      } else {
        float s = 0.f;
        for (int i = 0; i < 4; ++i) s += (float)row[i];
        res = s * 0.25f;
      }
      actB[r * 256 + o] = (_Float16)res;
    }
  }
  // ---- heads: act6 lives in actB cols 0..127 ----
  for (int t = tid; t < 256; t += 128) bias[t] = (t < 128) ? pib1[t] : 0.f;
  __syncthreads();
  gemm_t<1, 1>(actB, 256, wf + OFF_PI1, bias, 4, 8, actA, 256, nullptr, 0, nullptr, nullptr, 0, nullptr, 0);
  __syncthreads();
  for (int t = tid; t < 256; t += 128) bias[t] = (t < 81) ? pib2[t] : 0.f;
  __syncthreads();
  gemm_t<2, 1>(actA, 256, wf + OFF_PI2, bias, 4, 6, nullptr, 0, piF, 96, nullptr, nullptr, 0, nullptr, 0);
  __syncthreads();
  for (int t = tid; t < 256; t += 128) bias[t] = (t < 128) ? vb1[t] : 0.f;
  __syncthreads();
  gemm_t<1, 1>(actB, 256, wf + OFF_V1, bias, 4, 8, actA, 256, nullptr, 0, nullptr, nullptr, 0, nullptr, 0);
  __syncthreads();
  if (tid < 16) {
    float s = vb2[0];
    for (int j = 0; j < 128; ++j) s += (float)actA[tid * 256 + j] * vw2[j];
    out[(long)BATCH * 81 + bbase + tid] = tanhf(s);
  }
  __syncthreads();
  for (int t = tid; t < 256; t += 128) bias[t] = (t < 128) ? sdb1[t] : 0.f;
  __syncthreads();
  gemm_t<1, 1>(actB, 256, wf + OFF_SD1, bias, 4, 8, actA, 256, nullptr, 0, nullptr, nullptr, 0, nullptr, 0);
  __syncthreads();
  for (int t = tid; t < 256; t += 128) bias[t] = (t < 31) ? sdb2[t] : 0.f;
  __syncthreads();
  gemm_t<2, 1>(actA, 256, wf + OFF_SD2, bias, 4, 2, nullptr, 0, sdF, 32, nullptr, nullptr, 0, nullptr, 0);
  __syncthreads();
  // ---- masked log_softmax(pi, 81) / log_softmax(sd, 31) ----
  if (tid < 16) {
    long b = bbase + tid;
    const int* va = valid + b * 81;
    const float* pr = piF + tid * 96;
    float mx = -3.4e38f;
    for (int j = 0; j < 81; ++j) { float x = va[j] ? pr[j] : -1e8f; mx = fmaxf(mx, x); }
    float s = 0.f;
    for (int j = 0; j < 81; ++j) { float x = va[j] ? pr[j] : -1e8f; s += expf(x - mx); }
    float lse = mx + logf(s);
    float* po = out + b * 81;
    for (int j = 0; j < 81; ++j) { float x = va[j] ? pr[j] : -1e8f; po[j] = x - lse; }
  } else if (tid < 32) {
    int r = tid - 16;
    long b = bbase + r;
    const float* sr = sdF + r * 32;
    float mx = -3.4e38f;
    for (int j = 0; j < 31; ++j) mx = fmaxf(mx, sr[j]);
    float s = 0.f;
    for (int j = 0; j < 31; ++j) s += expf(sr[j] - mx);
    float lse = mx + logf(s);
    float* so = out + (long)BATCH * 82 + b * 31;
    for (int j = 0; j < 31; ++j) so[j] = sr[j] - lse;
  }
}

extern "C" void kernel_launch(void* const* d_in, const int* in_sizes, int n_in,
                              void* d_out, int out_size, void* d_ws, size_t ws_size,
                              hipStream_t stream) {
  _Float16* wf = (_Float16*)d_ws;
  _Float16* y3 = (_Float16*)((char*)d_ws + WF_BYTES);

  auto cv = [&](int idx, int N, int K, int Np, int Kp, int noff, int koff, int off) {
    int total = Np * Kp;
    wconv<<<dim3((total + 255) / 256), dim3(256), 0, stream>>>(
        (const float*)d_in[idx], wf + off, N, K, Np, Kp, noff, koff);
  };
  cv(2,  256, 56,  256, 64,  0, 0,  OFF_W1A);
  cv(8,  256, 256, 256, 256, 0, 0,  OFF_W1B);
  cv(10, 248, 224, 256, 256, 8, 32, OFF_GP1);
  cv(12, 128, 256, 128, 256, 0, 0,  OFF_W3);
  cv(14, 256, 704, 256, 704, 0, 0,  OFF_W4);
  cv(16, 248, 240, 256, 256, 8, 16, OFF_GP4);
  cv(18, 128, 256, 128, 256, 0, 0,  OFF_W5);
  cv(20, 120, 112, 128, 128, 8, 16, OFF_GP5);
  cv(22, 128, 128, 128, 128, 0, 0,  OFF_PI1);
  cv(24, 81,  128, 96,  128, 0, 0,  OFF_PI2);
  cv(26, 128, 128, 128, 128, 0, 0,  OFF_V1);
  cv(30, 128, 128, 128, 128, 0, 0,  OFF_SD1);
  cv(32, 31,  128, 32,  128, 0, 0,  OFF_SD2);

  stageA<<<dim3((BATCH * VD) / 32), dim3(128), 0, stream>>>(
      (const float*)d_in[0], wf,
      (const float*)d_in[3], (const float*)d_in[9],
      (const float*)d_in[11], (const float*)d_in[13],
      (const float*)d_in[4], (const float*)d_in[5],
      (const float*)d_in[6], (const float*)d_in[7],
      y3);

  stageB<<<dim3(BATCH / 16), dim3(128), 0, stream>>>(
      y3, wf,
      (const float*)d_in[15], (const float*)d_in[17],
      (const float*)d_in[19], (const float*)d_in[21],
      (const float*)d_in[23], (const float*)d_in[25],
      (const float*)d_in[27], (const float*)d_in[28], (const float*)d_in[29],
      (const float*)d_in[31], (const float*)d_in[33],
      (const int*)d_in[1],
      (float*)d_out);
}
